// TBD_26671746908279
// MI455X (gfx1250) — compile-verified, matrix operand order checked by reading, not run on hardware
//
#include <hip/hip_runtime.h>
#include <hip/hip_bf16.h>
#include <math.h>

// ---------------------------------------------------------------------------
// Temporal GNN (2-layer SAGE per block + MFGRU) for MI455X / gfx1250.
// GEMMs: v_wmma_f32_16x16x32_bf16, 4-way n-tile register blocking per wave,
// B strip staged in LDS (conflict-free ds_load_b128) and shared by all 4
// waves of the workgroup. Activations bf16; segment-means fp32 via
// global_atomic_add_f32.
// ---------------------------------------------------------------------------

#define N_NODES 50000      // 50000 = 16 * 3125  (M is WMMA-tile aligned)
#define E_EDGES 400000
#define T_BLOCKS 4
#define F_IN 128
#define TD 32
#define HID 256
#define D_OUT 128
#define SD 16
#define KBLK 256           // K-phase staged in LDS per iteration

typedef __bf16 bf16;
typedef bf16  v16bf __attribute__((ext_vector_type(16)));
typedef bf16  v8bf  __attribute__((ext_vector_type(8)));
typedef float v8f   __attribute__((ext_vector_type(8)));

struct Raw32 { uint4 lo, hi; };

__device__ __forceinline__ void atomAddF(float* p, float v) {
  __hip_atomic_fetch_add(p, v, __ATOMIC_RELAXED, __HIP_MEMORY_SCOPE_AGENT);
}

// A fragment (16x32 bf16, row-major MxK) per ISA 7.12.2:
//   lane<16 : row = lane,    holds K = k0+0..7  and k0+16..23
//   lane>=16: row = lane-16, holds K = k0+8..15 and k0+24..31
__device__ __forceinline__ v16bf load_frag(const bf16* __restrict__ base, int ld,
                                           int rc, int k0, int lane) {
  const bf16* p = base + (size_t)(rc + (lane & 15)) * ld + (k0 + ((lane >> 4) << 3));
  Raw32 r;
  r.lo = *(const uint4*)(p);
  r.hi = *(const uint4*)(p + 16);
  return __builtin_bit_cast(v16bf, r);
}

// Generic WMMA GEMM: out[M,Nout] = act(A[M,K] @ Wt[Nout,K]^T + bias)
// act: 0 none, 1 relu, 2 sigmoid, 3 tanh. Writes fp32 and/or bf16 outputs.
// 4 waves / block; each wave computes a 16x64 strip (4 n-tiles) so every A
// fragment feeds 4 WMMAs. The 64-row B strip for the current 256-wide K
// phase lives in LDS (row pad +8 bf16 -> bank advance 4/row: conflict-free).
// gridDim.y = Nout/64. Dynamic LDS: 64*(KBLK+8)*2 bytes.
__global__ void wmma_gemm(const bf16* __restrict__ A, int lda,
                          const bf16* __restrict__ Wt, int ldw,
                          const float* __restrict__ bias,
                          float* __restrict__ outF, int ldoF,
                          bf16* __restrict__ outB, int ldoB,
                          int M, int K, int act) {
  extern __shared__ char smemRaw[];
  bf16* Bs = (bf16*)smemRaw;
  const int  ldb   = KBLK + 8;
  const int  lane  = threadIdx.x & 31;
  const int  wave  = threadIdx.x >> 5;
  const int  mTile = blockIdx.x * 4 + wave;
  const bool active = (mTile * 16 < M);       // wave-uniform
  const int  m0 = mTile * 16;
  const int  n0 = blockIdx.y * 64;

  v8f acc[4] = {{}, {}, {}, {}};
  for (int kb = 0; kb < K; kb += KBLK) {
    const int kc = (K - kb < KBLK) ? (K - kb) : KBLK;
    // cooperative stage: Wt[n0 : n0+64, kb : kb+kc] -> LDS (uint4 granules)
    const int rowChunks = kc >> 3;
    for (int idx = threadIdx.x; idx < 64 * rowChunks; idx += (int)blockDim.x) {
      int row = idx / rowChunks, c8 = (idx % rowChunks) << 3;
      *(uint4*)(Bs + row * ldb + c8) =
          *(const uint4*)(Wt + (size_t)(n0 + row) * ldw + kb + c8);
    }
    __syncthreads();
    if (active) {
      for (int k0 = 0; k0 < kc; k0 += 32) {
        if (kb + k0 + 32 < K)  // near-cache streaming hint for A
          __builtin_prefetch(A + (size_t)(m0 + (lane & 15)) * lda + kb + k0 + 32, 0, 3);
        v16bf a = load_frag(A, lda, m0, kb + k0, lane);
#pragma unroll
        for (int j = 0; j < 4; ++j) {
          const bf16* p = Bs + (j * 16 + (lane & 15)) * ldb + (k0 + ((lane >> 4) << 3));
          Raw32 rr;
          rr.lo = *(const uint4*)(p);
          rr.hi = *(const uint4*)(p + 16);
          v16bf b = __builtin_bit_cast(v16bf, rr);
          acc[j] = __builtin_amdgcn_wmma_f32_16x16x32_bf16(false, a, false, b,
                                                           (short)0, acc[j], false, false);
        }
      }
    }
    __syncthreads();
  }
  if (!active) return;

  // C/D layout: col = lane&15, rows = (lane>=16 ? 8 : 0) + vgpr index
  const int rbase = m0 + ((lane >> 4) << 3);
#pragma unroll
  for (int j = 0; j < 4; ++j) {
    const int col  = n0 + j * 16 + (lane & 15);
    const float bb = bias ? bias[col] : 0.0f;
#pragma unroll
    for (int r = 0; r < 8; ++r) {
      float v = acc[j][r] + bb;
      if (act == 1)      v = fmaxf(v, 0.0f);
      else if (act == 2) v = 1.0f / (1.0f + __expf(-v));
      else if (act == 3) v = tanhf(v);
      if (outF) outF[(size_t)(rbase + r) * ldoF + col] = v;
      if (outB) outB[(size_t)(rbase + r) * ldoB + col] = (bf16)v;
    }
  }
}

// ---------------------------------------------------------------------------
// One-time weight conversion: bf16, transposed to [Nout, Kpad], K zero-padded.
__global__ void prep_weights(const float* __restrict__ Wself1, const float* __restrict__ Wneigh1,
                             const float* __restrict__ Wself2, const float* __restrict__ Wneigh2,
                             const float* __restrict__ Wr, const float* __restrict__ Wz,
                             const float* __restrict__ Wc, const float* __restrict__ Wf,
                             const float* __restrict__ br, const float* __restrict__ bz,
                             bf16* __restrict__ w1, bf16* __restrict__ w2,
                             bf16* __restrict__ wrz, bf16* __restrict__ wc,
                             bf16* __restrict__ wf, float* __restrict__ brz) {
  int idx = blockIdx.x * blockDim.x + threadIdx.x;
  const int s1 = 256 * 320, s2 = 128 * 512, s3 = 256 * 288,
            s4 = 128 * 288, s5 = 128 * 160, s6 = 256;
  if (idx < s1) {                              // layer1: [self || neigh], K=320
    int r = idx / 320, k = idx % 320;
    w1[idx] = (bf16)(k < 160 ? Wself1[k * 256 + r] : Wneigh1[(k - 160) * 256 + r]);
  } else if ((idx -= s1) < s2) {               // layer2: K=512
    int r = idx / 512, k = idx % 512;
    w2[idx] = (bf16)(k < 256 ? Wself2[k * 128 + r] : Wneigh2[(k - 256) * 128 + r]);
  } else if ((idx -= s2) < s3) {               // fused r||z gate: 256 out, K 272->288
    int r = idx / 288, k = idx % 288;
    wrz[idx] = (bf16)(k >= 272 ? 0.0f
                               : (r < 128 ? Wr[k * 128 + r] : Wz[k * 128 + (r - 128)]));
  } else if ((idx -= s3) < s4) {               // c gate: K 272->288
    int r = idx / 288, k = idx % 288;
    wc[idx] = (bf16)(k >= 272 ? 0.0f : Wc[k * 128 + r]);
  } else if ((idx -= s4) < s5) {               // f gate: K 144->160
    int r = idx / 160, k = idx % 160;
    wf[idx] = (bf16)(k >= 144 ? 0.0f : Wf[k * 128 + r]);
  } else if ((idx -= s5) < s6) {
    brz[idx] = (idx < 128) ? br[idx] : bz[idx - 128];
  }
}

__global__ void zero_f32(float* __restrict__ p, long n) {
  long i = (long)blockIdx.x * blockDim.x + threadIdx.x;
  if (i < n) p[i] = 0.0f;
}

// h0 = [feats[t] || time_table[t+1]] as bf16 rows of bufA1 (ld 320)
__global__ void build_h0(const float* __restrict__ featsT, const float* __restrict__ timeRow,
                         bf16* __restrict__ bufA1) {
  long i = (long)blockIdx.x * blockDim.x + threadIdx.x;
  if (i >= (long)N_NODES * 160) return;
  int n = (int)(i / 160), c = (int)(i % 160);
  float v = (c < 128) ? featsT[(size_t)n * 128 + c] : timeRow[c - 128];
  bufA1[(size_t)n * 320 + c] = (bf16)v;
}

__global__ void deg_count(const int* __restrict__ dst, float* __restrict__ deg) {
  int e = blockIdx.x * blockDim.x + threadIdx.x;
  if (e >= E_EDGES) return;
  atomAddF(&deg[dst[e]], 1.0f);
}

// agg[dst] += A[src] (fp32 accumulate of bf16 messages), 8 cols per thread.
__global__ void scatter_rows(const int* __restrict__ src, const int* __restrict__ dst,
                             const bf16* __restrict__ A, int lda,
                             float* __restrict__ agg, int K) {
  const int chunks = K >> 3;
  long i = (long)blockIdx.x * blockDim.x + threadIdx.x;
  if (i >= (long)E_EDGES * chunks) return;
  int e = (int)(i / chunks), c = (int)(i % chunks) << 3;
  int s = src[e], d = dst[e];
  v8bf vals = __builtin_bit_cast(v8bf, *(const uint4*)(A + (size_t)s * lda + c));
  float* q = agg + (size_t)d * K + c;
#pragma unroll
  for (int j = 0; j < 8; ++j) atomAddF(q + j, (float)vals[j]);
}

// mean over in-degree (>=1), converted to bf16 into column slice of dest buffer
__global__ void mean_to_bf(const float* __restrict__ agg, const float* __restrict__ deg,
                           bf16* __restrict__ out, int ldo, int colOff, int K) {
  long i = (long)blockIdx.x * blockDim.x + threadIdx.x;
  if (i >= (long)N_NODES * K) return;
  int n = (int)(i / K), c = (int)(i % K);
  float d = fmaxf(deg[n], 1.0f);
  out[(size_t)n * ldo + colOff + c] = (bf16)(agg[i] / d);
}

// combG = [x || h_prev || struct_feat || 0pad]  (N x 288, bf16)
__global__ void gru_combG(const bf16* __restrict__ xbf, const float* __restrict__ hprev,
                          const float* __restrict__ deg, const float* __restrict__ Wse,
                          const float* __restrict__ bse, bf16* __restrict__ combG) {
  long i = (long)blockIdx.x * blockDim.x + threadIdx.x;
  if (i >= (long)N_NODES * 288) return;
  int n = (int)(i / 288), c = (int)(i % 288);
  if (c < 128) { combG[i] = xbf[(size_t)n * 128 + c]; return; }
  float v;
  if (c < 256)      v = hprev ? hprev[(size_t)n * 128 + (c - 128)] : 0.0f;
  else if (c < 272) { int j = c - 256; v = fmaxf(deg[n] * Wse[j] + bse[j], 0.0f); }
  else              v = 0.0f;
  combG[i] = (bf16)v;
}

// combF = [x || struct_feat || 0pad]  (N x 160, bf16)
__global__ void gru_combF(const bf16* __restrict__ xbf, const float* __restrict__ deg,
                          const float* __restrict__ Wse, const float* __restrict__ bse,
                          bf16* __restrict__ combF) {
  long i = (long)blockIdx.x * blockDim.x + threadIdx.x;
  if (i >= (long)N_NODES * 160) return;
  int n = (int)(i / 160), c = (int)(i % 160);
  if (c < 128) { combF[i] = xbf[(size_t)n * 128 + c]; return; }
  float v = 0.0f;
  if (c < 144) { int j = c - 128; v = fmaxf(deg[n] * Wse[j] + bse[j], 0.0f); }
  combF[i] = (bf16)v;
}

// overwrite middle 128 cols of combG with r * h_prev (r already sigmoid'd)
__global__ void gru_mid(const float* __restrict__ rz, const float* __restrict__ hprev,
                        bf16* __restrict__ combG) {
  long i = (long)blockIdx.x * blockDim.x + threadIdx.x;
  if (i >= (long)N_NODES * 128) return;
  int n = (int)(i / 128), j = (int)(i % 128);
  float r  = rz[(size_t)n * 256 + j];
  float hp = hprev ? hprev[i] : 0.0f;
  combG[(size_t)n * 288 + 128 + j] = (bf16)(r * hp);
}

// h_new = fg * ((1-z)*h_prev + z*c)    (z sigmoid'd, c tanh'd, fg sigmoid'd)
__global__ void gru_final(const float* __restrict__ rz, const float* __restrict__ cbuf,
                          const float* __restrict__ fg, const float* __restrict__ hprev,
                          float* __restrict__ outT) {
  long i = (long)blockIdx.x * blockDim.x + threadIdx.x;
  if (i >= (long)N_NODES * 128) return;
  int n = (int)(i / 128), j = (int)(i % 128);
  float z  = rz[(size_t)n * 256 + 128 + j];
  float hp = hprev ? hprev[i] : 0.0f;
  outT[i]  = fg[i] * ((1.0f - z) * hp + z * cbuf[i]);
}

// ---------------------------------------------------------------------------
extern "C" void kernel_launch(void* const* d_in, const int* in_sizes, int n_in,
                              void* d_out, int out_size, void* d_ws, size_t ws_size,
                              hipStream_t stream) {
  const float* feats  = (const float*)d_in[0];
  const int*   src    = (const int*)  d_in[1];
  const int*   dst    = (const int*)  d_in[2];
  const float* tt     = (const float*)d_in[3];
  const float* Wself1 = (const float*)d_in[4];
  const float* Wneigh1= (const float*)d_in[5];
  const float* b1     = (const float*)d_in[6];
  const float* Wself2 = (const float*)d_in[7];
  const float* Wneigh2= (const float*)d_in[8];
  const float* b2     = (const float*)d_in[9];
  const float* Wse    = (const float*)d_in[10];
  const float* bse    = (const float*)d_in[11];
  const float* Wr     = (const float*)d_in[12];
  const float* br     = (const float*)d_in[13];
  const float* Wz     = (const float*)d_in[14];
  const float* bz     = (const float*)d_in[15];
  const float* Wc     = (const float*)d_in[16];
  const float* bc     = (const float*)d_in[17];
  const float* Wf     = (const float*)d_in[18];
  const float* bfb    = (const float*)d_in[19];
  float* out = (float*)d_out;

  // ---- workspace layout (256B aligned slices) ----
  char* wsb = (char*)d_ws;
  size_t off = 0;
  auto take = [&](size_t bytes) -> char* {
    size_t o = (off + 255) & ~(size_t)255; off = o + bytes; return wsb + o;
  };
  const size_t N = N_NODES;
  bf16*  w1    = (bf16*) take(256 * 320 * 2);
  bf16*  w2    = (bf16*) take(128 * 512 * 2);
  bf16*  wrz   = (bf16*) take(256 * 288 * 2);
  bf16*  wc    = (bf16*) take(128 * 288 * 2);
  bf16*  wf    = (bf16*) take(128 * 160 * 2);
  float* brz   = (float*)take(256 * 4);
  bf16*  bufA1 = (bf16*) take(N * 320 * 2);   // h0||mean0 ; reused as fp32 c-gate
  bf16*  bufA2 = (bf16*) take(N * 512 * 2);   // h1||mean1
  float* agg   = (float*)take(N * 256 * 4);   // segment sums ; reused as rz
  float* deg   = (float*)take(N * 4);
  bf16*  xbf   = (bf16*) take(N * 128 * 2);   // block representation (bf16)
  bf16*  combG = (bf16*) take(N * 288 * 2);
  bf16*  combF = (bf16*) take(N * 160 * 2);
  float* fg    = (float*)take(N * 128 * 4);
  float* cbuf  = (float*)bufA1;               // alias: bufA1 dead during GRU
  float* rz    = agg;                         // alias: agg dead after mean1

  const int TPB = 256;
  auto g1d = [&](long total) { return dim3((unsigned)((total + TPB - 1) / TPB)); };

  // weights -> bf16 transposed/padded (once per launch)
  hipLaunchKernelGGL(prep_weights, g1d(278784), dim3(TPB), 0, stream,
                     Wself1, Wneigh1, Wself2, Wneigh2, Wr, Wz, Wc, Wf, br, bz,
                     w1, w2, wrz, wc, wf, brz);

  const dim3     gemmBlk(128);                 // 4 waves, one 16x64 strip each
  const unsigned mGrid = (3125 + 3) / 4;       // 50000/16 m-tiles, 4 per block
  const size_t   ldsB  = (size_t)64 * (KBLK + 8) * 2;  // 33.8 KB B-strip tile

  for (int t = 0; t < T_BLOCKS; ++t) {
    const float* hprev = t ? out + (size_t)(t - 1) * N * 128 : nullptr;
    float*       outT  = out + (size_t)t * N * 128;
    const int*   srcT  = src + (size_t)t * E_EDGES;
    const int*   dstT  = dst + (size_t)t * E_EDGES;

    // ---- SAGE layer 1 ----
    hipLaunchKernelGGL(zero_f32, g1d(N * 160), dim3(TPB), 0, stream, agg, (long)(N * 160));
    hipLaunchKernelGGL(zero_f32, g1d(N),       dim3(TPB), 0, stream, deg, (long)N);
    hipLaunchKernelGGL(build_h0, g1d(N * 160), dim3(TPB), 0, stream,
                       feats + (size_t)t * N * 128, tt + (size_t)(t + 1) * TD, bufA1);
    hipLaunchKernelGGL(deg_count, g1d(E_EDGES), dim3(TPB), 0, stream, dstT, deg);
    hipLaunchKernelGGL(scatter_rows, g1d((long)E_EDGES * 20), dim3(TPB), 0, stream,
                       srcT, dstT, bufA1, 320, agg, 160);
    hipLaunchKernelGGL(mean_to_bf, g1d(N * 160), dim3(TPB), 0, stream,
                       agg, deg, bufA1, 320, 160, 160);
    // h1 = relu([h0||mean0] @ W1cat + b1) -> bf16 cols 0:256 of bufA2
    hipLaunchKernelGGL(wmma_gemm, dim3(mGrid, 4), gemmBlk, ldsB, stream,
                       bufA1, 320, w1, 320, b1, (float*)nullptr, 0, bufA2, 512,
                       (int)N, 320, 1);

    // ---- SAGE layer 2 ----
    hipLaunchKernelGGL(zero_f32, g1d(N * 256), dim3(TPB), 0, stream, agg, (long)(N * 256));
    hipLaunchKernelGGL(scatter_rows, g1d((long)E_EDGES * 32), dim3(TPB), 0, stream,
                       srcT, dstT, bufA2, 512, agg, 256);
    hipLaunchKernelGGL(mean_to_bf, g1d(N * 256), dim3(TPB), 0, stream,
                       agg, deg, bufA2, 512, 256, 256);
    // x = [h1||mean1] @ W2cat + b2 -> bf16
    hipLaunchKernelGGL(wmma_gemm, dim3(mGrid, 2), gemmBlk, ldsB, stream,
                       bufA2, 512, w2, 512, b2, (float*)nullptr, 0, xbf, 128,
                       (int)N, 512, 0);

    // ---- MFGRU step ----
    hipLaunchKernelGGL(gru_combG, g1d(N * 288), dim3(TPB), 0, stream,
                       xbf, hprev, deg, Wse, bse, combG);
    hipLaunchKernelGGL(gru_combF, g1d(N * 160), dim3(TPB), 0, stream,
                       xbf, deg, Wse, bse, combF);
    // fused r||z = sigmoid(combG @ Wrz + brz)   (fp32, into rz==agg)
    hipLaunchKernelGGL(wmma_gemm, dim3(mGrid, 4), gemmBlk, ldsB, stream,
                       combG, 288, wrz, 288, brz, rz, 256, (bf16*)nullptr, 0,
                       (int)N, 288, 2);
    hipLaunchKernelGGL(gru_mid, g1d(N * 128), dim3(TPB), 0, stream, rz, hprev, combG);
    // c = tanh([x||r*h||sf] @ Wc + bc)
    hipLaunchKernelGGL(wmma_gemm, dim3(mGrid, 2), gemmBlk, ldsB, stream,
                       combG, 288, wc, 288, bc, cbuf, 128, (bf16*)nullptr, 0,
                       (int)N, 288, 3);
    // fg = sigmoid([x||sf] @ Wf + bf)
    hipLaunchKernelGGL(wmma_gemm, dim3(mGrid, 2), gemmBlk, ldsB, stream,
                       combF, 160, wf, 160, bfb, fg, 128, (bf16*)nullptr, 0,
                       (int)N, 160, 2);
    hipLaunchKernelGGL(gru_final, g1d(N * 128), dim3(TPB), 0, stream,
                       rz, cbuf, fg, hprev, outT);
  }
}
